// EncoderGRUODE_35699768165107
// MI455X (gfx1250) — matrix operand order repaired and checked by reading, not compiled, not run on hardware
//
#include <hip/hip_runtime.h>

// Problem dimensions (fixed by the reference).
#define BB 512   // batch
#define TT 256   // time steps
#define DD 64    // input dim
#define HH 256   // hidden dim (2*LATENT)
#define H3 768   // 3*H

// LDS tile pitches (padded to dodge bank conflicts; rows stay 16B-aligned).
#define PITCH  264   // bf16 tile row pitch in shorts (528B = 33*16)
#define FPITCH 260   // f32 tile row pitch in floats

typedef __attribute__((ext_vector_type(16))) __bf16        v16bf;
typedef __attribute__((ext_vector_type(8)))  float         v8f;
typedef __attribute__((ext_vector_type(4)))  unsigned int  u32x4;

__device__ __forceinline__ unsigned short f32_to_bf16(float f) {
  unsigned int u = __float_as_uint(f);
  u += 0x7FFFu + ((u >> 16) & 1u);           // round-to-nearest-even
  return (unsigned short)(u >> 16);
}

__device__ __forceinline__ float bf16_to_f32(unsigned short h) {
  return __uint_as_float(((unsigned int)h) << 16);
}

// CDNA5 has a hardware transcendental tanh (v_tanh_f32). Use the builtin when
// this clang exposes it (either spelling); otherwise fall back to libm tanhf.
__device__ __forceinline__ float fast_tanh(float x) {
#if __has_builtin(__builtin_amdgcn_tanhf)
  return __builtin_amdgcn_tanhf(x);
#elif __has_builtin(__builtin_amdgcn_tanh_f32)
  return __builtin_amdgcn_tanh_f32(x);
#else
  return tanhf(x);
#endif
}

// sigmoid(x) = 0.5*tanh(x/2) + 0.5 : one TRANS op + two FMAs on the serial
// inter-barrier path (vs the branchy ocml exp expansion).
__device__ __forceinline__ float sigm(float x) {
  return __builtin_fmaf(0.5f, fast_tanh(0.5f * x), 0.5f);
}

// A-fragment (16x32 bf16) from row-major [M][ldk] source (global or LDS):
// lane l: row = m0 + (l&15); K = kb..kb+15, kb = k0 + (l>>4)*16
// -> two consecutive 16B loads (global_load_b128 or ds_load_b128).
__device__ __forceinline__ v16bf load_frag_a(const unsigned short* base, int ldk,
                                             int m0, int k0, int lane) {
  int row = m0 + (lane & 15);
  int kb  = k0 + ((lane >> 4) << 4);
  const u32x4* p = reinterpret_cast<const u32x4*>(base + (size_t)row * ldk + kb);
  union { v16bf v; u32x4 u[2]; } f;
  f.u[0] = p[0];
  f.u[1] = p[1];
  return f.v;
}

// B-fragment (32x16 bf16) of W stored row-major [N][ldk] (GEMM computes in @ W^T):
// lane l: col n = n0 + (l&15); K = kb..kb+15, kb = k0 + (l>>4)*16 -> 32 contiguous bytes.
__device__ __forceinline__ v16bf load_frag_b(const unsigned short* Wb, int ldk,
                                             int n0, int k0, int lane) {
  int n  = n0 + (lane & 15);
  int kb = k0 + ((lane >> 4) << 4);
  const u32x4* p = reinterpret_cast<const u32x4*>(Wb + (size_t)n * ldk + kb);
  union { v16bf v; u32x4 u[2]; } f;
  f.u[0] = p[0];
  f.u[1] = p[1];
  return f.v;
}

#define WMMA_BF16(A, Bf, C) \
  __builtin_amdgcn_wmma_f32_16x16x32_bf16(false, (A), false, (Bf), (short)0, (C), false, false)

// Compiler-level memory barrier: placed once per t-iteration so LICM cannot
// hoist the loop-invariant weight fragment loads out of the 256-step loop
// (hoisting ~370 VGPRs of invariant B-fragments caused scratch spill/reload
// in the hot loop). Weights instead re-load each step from L2 (they are
// ~736KB, fully resident in the 192MB L2 and shared by all lanes).
#define NO_LICM() asm volatile("" ::: "memory")

// ---------------- prologue: f32 -> bf16 weight/x conversion ----------------

__global__ void k_cvt_bf16(const float* __restrict__ src,
                           unsigned short* __restrict__ dst, int n) {
  int i = blockIdx.x * blockDim.x + threadIdx.x;
  int stride = gridDim.x * blockDim.x;
  for (; i < n; i += stride) dst[i] = f32_to_bf16(src[i]);
}

// ---------------- persistent scan kernel ----------------
// One block per 16 batch rows; 512 threads = 16 wave32s; wave w owns output
// columns [16w, 16w+16). All recurrent state lives in LDS; weights stream
// from L2 as bf16 B-fragments. The whole T=256 scan runs in one launch with
// only workgroup barriers between intra-step phases.

__global__ __launch_bounds__(512) void k_scan(
    const unsigned short* __restrict__ w1b,
    const unsigned short* __restrict__ w2b,
    const unsigned short* __restrict__ whhb,
    const unsigned short* __restrict__ wihb,
    const unsigned short* __restrict__ woutb,
    const unsigned short* __restrict__ xb,
    const float* __restrict__ tps,
    const float* __restrict__ mask,
    const float* __restrict__ b1,
    const float* __restrict__ b2,
    const float* __restrict__ b_ih,
    const float* __restrict__ b_hh,
    const float* __restrict__ b_out,
    float* __restrict__ out) {
  __shared__ __align__(16) unsigned short s_h_bf[16 * PITCH];     // h state, bf16
  __shared__ __align__(16) unsigned short s_a1_bf[16 * PITCH];    // tanh layer-1 act
  __shared__ __align__(16) unsigned short s_hode_bf[16 * PITCH];  // h after ODE step
  __shared__ __align__(16) unsigned short s_hlast_bf[16 * PITCH]; // state at last valid t
  __shared__ __align__(16) float          s_h_f32[16 * FPITCH];   // h state, f32 master
  __shared__ float s_seen[16];                                    // r_fill per row

  int tid  = threadIdx.x;
  int lane = tid & 31;
  int w    = tid >> 5;                 // wave id: 0..15
  int n0   = w * 16;                   // output column tile base
  int m0   = blockIdx.x * 16;          // global batch row base
  int n    = n0 + (lane & 15);         // this lane's output column
  int rloc = (lane >> 4) << 3;         // C-fragment row base within tile (0 or 8)

  // init state
  for (int i = tid; i < 16 * PITCH; i += 512) { s_h_bf[i] = 0; s_hlast_bf[i] = 0; }
  for (int i = tid; i < 16 * FPITCH; i += 512) s_h_f32[i] = 0.0f;
  if (tid < 16) s_seen[tid] = 0.0f;
  __syncthreads();

  // per-lane biases (loaded once; SSA values stay in registers)
  float bias1   = b1[n];
  float bias2   = b2[n];
  float br_bias = b_ih[n] + b_hh[n];
  float bz_bias = b_ih[HH + n] + b_hh[HH + n];
  float bihn    = b_ih[2 * HH + n];
  float bhhn    = b_hh[2 * HH + n];

#pragma unroll 1
  for (int t = 0; t < TT; ++t) {
    NO_LICM();  // keep weight loads inside the loop (see comment above)

    // ---- phase A: a1 = tanh(h @ w1^T + b1) ----
    {
      v8f acc = {};
      for (int k = 0; k < HH; k += 32) {
        v16bf a  = load_frag_a(s_h_bf, PITCH, 0, k, lane);
        v16bf bb = load_frag_b(w1b, HH, n0, k, lane);
        acc = WMMA_BF16(a, bb, acc);
      }
#pragma unroll
      for (int i = 0; i < 8; ++i)
        s_a1_bf[(rloc + i) * PITCH + n] = f32_to_bf16(fast_tanh(acc[i] + bias1));
    }
    __syncthreads();

    // ---- phase B: f = a1 @ w2^T + b2; Euler step + r_fill blend ----
    {
      v8f acc = {};
      for (int k = 0; k < HH; k += 32) {
        v16bf a  = load_frag_a(s_a1_bf, PITCH, 0, k, lane);
        v16bf bb = load_frag_b(w2b, HH, n0, k, lane);
        acc = WMMA_BF16(a, bb, acc);
      }
      float dt = (t == 0) ? 0.01f : (tps[t] - tps[t - 1]);
#pragma unroll
      for (int i = 0; i < 8; ++i) {
        int rl = rloc + i;
        float hp = s_h_f32[rl * FPITCH + n];
        float fv = acc[i] + bias2;
        float he = hp + dt * fv;
        float rr = s_seen[rl];
        float ho = he * rr + hp * (1.0f - rr);
        s_hode_bf[rl * PITCH + n] = f32_to_bf16(ho);
      }
    }
    __syncthreads();

    // ---- phase C: fused GRU gates + mask blend + last-state tracking ----
    {
      v8f acc_r = {}, acc_z = {}, gh_n = {}, gi_n = {};
      for (int k = 0; k < HH; k += 32) {
        v16bf a   = load_frag_a(s_hode_bf, PITCH, 0, k, lane);
        v16bf brf = load_frag_b(whhb, HH, n0,          k, lane);
        v16bf bzf = load_frag_b(whhb, HH, HH + n0,     k, lane);
        v16bf bnf = load_frag_b(whhb, HH, 2 * HH + n0, k, lane);
        acc_r = WMMA_BF16(a, brf, acc_r);
        acc_z = WMMA_BF16(a, bzf, acc_z);
        gh_n  = WMMA_BF16(a, bnf, gh_n);
      }
      const unsigned short* xrow = xb + (size_t)t * DD;  // row m at xrow + m*TT*DD
      for (int k = 0; k < DD; k += 32) {
        v16bf a   = load_frag_a(xrow, TT * DD, m0, k, lane);
        v16bf brf = load_frag_b(wihb, DD, n0,          k, lane);
        v16bf bzf = load_frag_b(wihb, DD, HH + n0,     k, lane);
        v16bf bnf = load_frag_b(wihb, DD, 2 * HH + n0, k, lane);
        acc_r = WMMA_BF16(a, brf, acc_r);
        acc_z = WMMA_BF16(a, bzf, acc_z);
        gi_n  = WMMA_BF16(a, bnf, gi_n);
      }
#pragma unroll
      for (int i = 0; i < 8; ++i) {
        int rl = rloc + i;
        float rg   = sigm(acc_r[i] + br_bias);
        float zg   = sigm(acc_z[i] + bz_bias);
        float ng   = fast_tanh((gi_n[i] + bihn) + rg * (gh_n[i] + bhhn));
        float ho   = bf16_to_f32(s_hode_bf[rl * PITCH + n]);
        float hrnn = (1.0f - zg) * ng + zg * ho;
        float mt   = mask[(size_t)(m0 + rl) * TT + t];
        float hv   = hrnn * mt + ho * (1.0f - mt);
        s_h_f32[rl * FPITCH + n] = hv;
        s_h_bf[rl * PITCH + n]   = f32_to_bf16(hv);
        if (mt > 0.5f) s_hlast_bf[rl * PITCH + n] = f32_to_bf16(hrnn);
      }
    }
    __syncthreads();

    // update r_fill "seen" flags for step t+1
    if (tid < 16) {
      float mt = mask[(size_t)(m0 + tid) * TT + t];
      if (mt > 0.5f) s_seen[tid] = 1.0f;
    }
    __syncthreads();
  }

  // ---- epilogue: out = h_last @ w_out^T + b_out ----
  {
    v8f acc = {};
    for (int k = 0; k < HH; k += 32) {
      v16bf a  = load_frag_a(s_hlast_bf, PITCH, 0, k, lane);
      v16bf bb = load_frag_b(woutb, HH, n0, k, lane);
      acc = WMMA_BF16(a, bb, acc);
    }
    float bias = b_out[n];
#pragma unroll
    for (int i = 0; i < 8; ++i)
      out[(size_t)(m0 + rloc + i) * HH + n] = acc[i] + bias;
  }
}

// ---------------- host ----------------

extern "C" void kernel_launch(void* const* d_in, const int* in_sizes, int n_in,
                              void* d_out, int out_size, void* d_ws, size_t ws_size,
                              hipStream_t stream) {
  (void)in_sizes; (void)n_in; (void)out_size; (void)ws_size;
  const float* x     = (const float*)d_in[0];   // (B,T,D)
  const float* tps   = (const float*)d_in[1];   // (T,)
  const float* mask  = (const float*)d_in[2];   // (B,T)
  const float* w_ih  = (const float*)d_in[3];   // (3H,D)
  const float* w_hh  = (const float*)d_in[4];   // (3H,H)
  const float* b_ih  = (const float*)d_in[5];
  const float* b_hh  = (const float*)d_in[6];
  const float* w1    = (const float*)d_in[7];   // (H,H)
  const float* b1    = (const float*)d_in[8];
  const float* w2    = (const float*)d_in[9];   // (H,H)
  const float* b2    = (const float*)d_in[10];
  const float* w_out = (const float*)d_in[11];  // (H,H)
  const float* b_out = (const float*)d_in[12];
  float* out = (float*)d_out;

  char* ws = (char*)d_ws;
  size_t off = 0;
  auto carve = [&](size_t bytes) -> char* {
    char* p = ws + off;
    off = (off + bytes + 255) & ~(size_t)255;
    return p;
  };
  unsigned short* w1b   = (unsigned short*)carve((size_t)HH * HH * 2);
  unsigned short* w2b   = (unsigned short*)carve((size_t)HH * HH * 2);
  unsigned short* whhb  = (unsigned short*)carve((size_t)H3 * HH * 2);
  unsigned short* wihb  = (unsigned short*)carve((size_t)H3 * DD * 2);
  unsigned short* woutb = (unsigned short*)carve((size_t)HH * HH * 2);
  unsigned short* xb    = (unsigned short*)carve((size_t)BB * TT * DD * 2);

  auto cvt = [&](const float* s, unsigned short* d, int nelem) {
    int blocks = (nelem + 255) / 256;
    if (blocks > 4096) blocks = 4096;
    k_cvt_bf16<<<blocks, 256, 0, stream>>>(s, d, nelem);
  };
  cvt(w1,    w1b,   HH * HH);
  cvt(w2,    w2b,   HH * HH);
  cvt(w_hh,  whhb,  H3 * HH);
  cvt(w_ih,  wihb,  H3 * DD);
  cvt(w_out, woutb, HH * HH);
  cvt(x,     xb,    BB * TT * DD);

  // One persistent block per 16 batch rows; 16 wave32s per block.
  k_scan<<<dim3(BB / 16), dim3(512), 0, stream>>>(
      w1b, w2b, whhb, wihb, woutb, xb, tps, mask,
      b1, b2, b_ih, b_hh, b_out, out);
}
